// PCMMtorchBaseModel_56882546868648
// MI455X (gfx1250) — matrix operand order repaired and checked by reading, not compiled, not run on hardware
//
#include <hip/hip_runtime.h>
#include <hip/hip_bf16.h>

// CDNA5 / gfx1250 HMM forward.
// Scaled-linear recursion: p' = 2^-e * ( exp(lp_t) ∘ (Texp^T · p) ), c_int += e,
// with e = exponent of the per-column sum (exact power-of-two renormalization).
// State mixing = two v_wmma_f32_16x16x32_f16 per step, 16 sequences batched as N.

typedef __attribute__((ext_vector_type(16))) _Float16 v16h;
typedef __attribute__((ext_vector_type(8)))  float    v8f;

#define HMM_K        32
#define HMM_N        1048576
#define HMM_NS       512
#define SEQ_PER_WAVE 16
#define NUM_SEQ      (HMM_N / HMM_NS)         /* 2048 */
#define NUM_WAVES    (NUM_SEQ / SEQ_PER_WAVE) /* 128 */

// ds_swizzle group-of-32: offset[14:10]=xor, [9:5]=or, [4:0]=and
#define SWZ_X1  0x041f
#define SWZ_X2  0x081f
#define SWZ_X4  0x101f
#define SWZ_X8  0x201f
#define SWZ_X16 0x401f

template <int OFF>
__device__ __forceinline__ float swz(float x) {
  return __int_as_float(__builtin_amdgcn_ds_swizzle(__float_as_int(x), OFF));
}

__device__ __forceinline__ float red_max32(float v) {
  v = fmaxf(v, swz<SWZ_X1>(v));
  v = fmaxf(v, swz<SWZ_X2>(v));
  v = fmaxf(v, swz<SWZ_X4>(v));
  v = fmaxf(v, swz<SWZ_X8>(v));
  v = fmaxf(v, swz<SWZ_X16>(v));
  return v;
}
__device__ __forceinline__ float red_sum32(float v) {
  v += swz<SWZ_X1>(v);
  v += swz<SWZ_X2>(v);
  v += swz<SWZ_X4>(v);
  v += swz<SWZ_X8>(v);
  v += swz<SWZ_X16>(v);
  return v;
}
__device__ __forceinline__ float red_sum16(float v) {
  v += swz<SWZ_X1>(v);
  v += swz<SWZ_X2>(v);
  v += swz<SWZ_X4>(v);
  v += swz<SWZ_X8>(v);
  return v;
}

__global__ __launch_bounds__(32)
void hmm_forward_wave(const float* __restrict__ log_pdf,
                      const float* __restrict__ pi,
                      const float* __restrict__ T,
                      float* __restrict__ partial)
{
  __shared__ _Float16 texp[HMM_K * HMM_K];   // Texp[i][j] = softmax row i of T
  __shared__ float    logpi_s[HMM_K];

  const int  lane  = threadIdx.x;            // 0..31 (wave32)
  const int  s_loc = lane & 15;
  const bool lo    = (lane < 16);
  const int  hi8   = lo ? 0 : 8;

  // ---- one-time: log_softmax(pi), softmax rows of T (to f16 LDS) ----
  {
    float p = pi[lane];
    float m = red_max32(p);
    float e = __expf(p - m);
    float s = red_sum32(e);
    logpi_s[lane] = p - m - __logf(s);
  }
  for (int i = 0; i < HMM_K; ++i) {
    float v = T[i * HMM_K + lane];
    float m = red_max32(v);
    float e = __expf(v - m);
    float s = red_sum32(e);
    texp[i * HMM_K + lane] = (_Float16)(e / s);
  }
  __syncthreads();

  // ---- A tiles: A_h = Texp^T rows j in [16h,16h+16), K = 32 input states ----
  // f16 A 16x32, contiguous k order used here (B packed to match): lane<16: row=lane, elems 0..15 -> K=0..15
  //                     lane>=16: row=lane-16, elems 0..15 -> K=16..31
  v16h A0, A1;
  {
    const int Mr = s_loc;
    const int kb = lo ? 0 : 16;
#pragma unroll
    for (int e = 0; e < 16; ++e) {
      int k = e + kb;                                  // input state i
      A0[e] = texp[k * HMM_K + Mr];                    // Texp[i][j=Mr]
      A1[e] = texp[k * HMM_K + 16 + Mr];               // Texp[i][j=16+Mr]
    }
  }

  // ---- emission base offsets: D tile h, vgpr r holds (j = 16h+r+hi8, s = s_loc) ----
  const int s_base = blockIdx.x * SEQ_PER_WAVE;
  unsigned off[2][8];
#pragma unroll
  for (int h = 0; h < 2; ++h)
#pragma unroll
    for (int r = 0; r < 8; ++r)
      off[h][r] = (unsigned)((16 * h + r + hi8) * HMM_N)
                + (unsigned)((s_base + s_loc) * HMM_NS);

  int   c_int = 0;         // exact power-of-two scale carry (per column, replicated)
  float mant  = 1.0f;      // mantissa of last column sum, in [0.5, 1)
  float d[2][8];           // D-layout f32 state (2 tiles x 8 vgprs)
  v16h  B;                 // f16 B-layout state fed to WMMA

  // Renormalize column s by 2^-e (exact), carry exponent, repack D->B layout.
  // B lane<16 needs K=0..15 (tile0); lane>=16 needs K=16..31 (tile1); half the
  // rows live on the opposite lane-half -> one SWAPX16 swizzle per m (pre-select
  // the outgoing value, since hi lanes must send tile0 and lo lanes tile1).
  auto norm_pack = [&]() {
    float s0  = (d[0][0] + d[0][1]) + (d[0][2] + d[0][3]);   // balanced tree:
    float s1  = (d[0][4] + d[0][5]) + (d[0][6] + d[0][7]);   // depth 4, not a
    float s2  = (d[1][0] + d[1][1]) + (d[1][2] + d[1][3]);   // 16-deep chain
    float s3  = (d[1][4] + d[1][5]) + (d[1][6] + d[1][7]);
    float s16 = (s0 + s1) + (s2 + s3);
    float tot = s16 + swz<SWZ_X16>(s16);     // full 32-state column sum, both halves
    int   e   = __builtin_amdgcn_frexp_expf(tot);    // tot = mant * 2^e
    mant      = __builtin_amdgcn_frexp_mantf(tot);
    c_int    += e;
#pragma unroll
    for (int m = 0; m < 8; ++m) {
      float send = lo ? d[1][m] : d[0][m];   // value the other half needs
      float x    = swz<SWZ_X16>(send);
      float blo  = lo ? d[0][m] : x;         // K=m       | K=16+m
      float bhi  = lo ? x : d[1][m];         // K=8+m     | K=24+m
      B[m]     = (_Float16)__builtin_amdgcn_ldexpf(blo, -e);
      B[m + 8] = (_Float16)__builtin_amdgcn_ldexpf(bhi, -e);
    }
  };

  // ---- t = 0: p0 = exp(lp0 + log_pi), normalized ----
#pragma unroll
  for (int h = 0; h < 2; ++h)
#pragma unroll
    for (int r = 0; r < 8; ++r) {
      float lpv = log_pdf[off[h][r]];
      float lpi = logpi_s[16 * h + r + hi8];
      d[h][r] = __expf(lpv + lpi);
    }
  norm_pack();

  // ---- recursion t = 1..Ns-1 ----
  const v8f czero = {0.f, 0.f, 0.f, 0.f, 0.f, 0.f, 0.f, 0.f};
#pragma unroll 4
  for (int t = 1; t < HMM_NS; ++t) {
    float em[2][8];                          // emissions: independent of chain
#pragma unroll
    for (int h = 0; h < 2; ++h)
#pragma unroll
      for (int r = 0; r < 8; ++r)
        em[h][r] = log_pdf[off[h][r] + t];

    if ((t & 31) == 1 && (t + 32) < HMM_NS) {    // pull next 128B lines early
#pragma unroll
      for (int h = 0; h < 2; ++h)
#pragma unroll
        for (int r = 0; r < 8; ++r)
          __builtin_prefetch(&log_pdf[off[h][r] + t + 32], 0, 1);
    }

    v8f d0 = __builtin_amdgcn_wmma_f32_16x16x32_f16(
        false, A0, false, B, (short)0, czero, false, false);
    v8f d1 = __builtin_amdgcn_wmma_f32_16x16x32_f16(
        false, A1, false, B, (short)0, czero, false, false);

#pragma unroll
    for (int r = 0; r < 8; ++r) {
      d[0][r] = d0[r] * __expf(em[0][r]);
      d[1][r] = d1[r] * __expf(em[1][r]);
    }
    norm_pack();
  }

  // logsumexp(alpha_T) per sequence = c_int*ln2 + log(mant)  (exact carry).
  float log_t = (float)c_int * 0.69314718055994530942f + __logf(mant);
  float wsum  = red_sum16(lo ? log_t : 0.f);   // lanes 0..15 hold the 16 seqs
  if (lane == 0) partial[blockIdx.x] = wsum;
}

__global__ __launch_bounds__(32)
void hmm_reduce(const float* __restrict__ partial, float* __restrict__ out, int n)
{
  float s = 0.f;
  for (int i = threadIdx.x; i < n; i += 32) s += partial[i];   // fixed order
  s = red_sum32(s);
  if (threadIdx.x == 0) out[0] = s;
}

extern "C" void kernel_launch(void* const* d_in, const int* in_sizes, int n_in,
                              void* d_out, int out_size, void* d_ws, size_t ws_size,
                              hipStream_t stream)
{
  (void)in_sizes; (void)n_in; (void)out_size; (void)ws_size;
  const float* log_pdf = (const float*)d_in[0];
  const float* pi      = (const float*)d_in[1];
  const float* T       = (const float*)d_in[2];
  float* partial = (float*)d_ws;

  hmm_forward_wave<<<NUM_WAVES, 32, 0, stream>>>(log_pdf, pi, T, partial);
  hmm_reduce<<<1, 32, 0, stream>>>(partial, (float*)d_out, NUM_WAVES);
}
